// SimpleMetaConvKALN_27874337751726
// MI455X (gfx1250) — matrix operand order repaired and checked by reading, not compiled, not run on hardware
//
#include <hip/hip_runtime.h>
#include <math.h>

// ---- problem constants (from reference) ----
#define BATCH   64
#define IN_DIM  64
#define OUT_DIM 128
#define NGROUPS 4
#define IN_G    16
#define OUT_G   32
#define HW      64
#define RCH     64     // GEMM basis channels: silu + P1 + P2 + P3 (P0-ones -> bias table)
#define TAPS    9

typedef float          v8f   __attribute__((ext_vector_type(8)));
typedef __bf16         v16bf __attribute__((ext_vector_type(16)));
typedef unsigned short v16us __attribute__((ext_vector_type(16)));
typedef unsigned short v8us  __attribute__((ext_vector_type(8)));

#define NWPACK (2 * TAPS * 2 * 32 * 16)   // 18,432 bf16 = 36,864 B
#define NWB    (9 * OUT_G)                // 288 f32 bias table

__device__ __forceinline__ unsigned short f2bf(float f) {
  unsigned u = __float_as_uint(f);
  u += 0x7fffu + ((u >> 16) & 1u);          // round-to-nearest-even
  return (unsigned short)(u >> 16);
}
__device__ __forceinline__ float silu(float x) { return x / (1.f + __expf(-x)); }

// monotone-increasing uint key for float ordering (works with atomicMin/Max u32)
__device__ __forceinline__ unsigned fkey(float f) {
  unsigned u = __float_as_uint(f);
  return (u & 0x80000000u) ? ~u : (u | 0x80000000u);
}
__device__ __forceinline__ float funkey(unsigned k) {
  unsigned u = (k & 0x80000000u) ? (k & 0x7fffffffu) : ~k;
  return __uint_as_float(u);
}
__device__ __forceinline__ v16bf mkfrag(v8us lo, v8us hi) {
  v16us a = __builtin_shufflevector(lo, hi, 0,1,2,3,4,5,6,7,8,9,10,11,12,13,14,15);
  return __builtin_bit_cast(v16bf, a);
}

// ---- kernel 1: min/max key init + weight repack (bf16 WMMA-B layout) + ones-bias table ----
// wpack[t][tap][s][lane][j]:  lane<16: N=lane, K=s*32+j ; lane>=16: N=lane-16, K=s*32+16+j
// GEMM channel K: 0-15 -> lw ch K (silu/base), 16-63 -> lw ch K+16 (P1,P2,P3 blocks).
// Wb[chh][cww][oc] = sum over taps valid for border case (chh,cww) of
//                    sum_c lw[oc][16+c][kh][kw]   (the P0-ones contribution)
__global__ void k_init(const float* __restrict__ lw, unsigned* __restrict__ mm,
                       unsigned short* __restrict__ wpack, float* __restrict__ wb) {
  int e = blockIdx.x * 256 + threadIdx.x;
  if (e < 8) mm[e] = (e & 1) ? 0u : 0xFFFFFFFFu;   // even = min key, odd = max key
  if (e < NWPACK) {
    int j    = e & 15;
    int lane = (e >> 4) & 31;
    int s    = (e >> 9) & 1;
    int tap  = (e / 1024) % 9;
    int t    = e / 9216;
    int K  = s * 32 + ((lane & 16) ? 16 : 0) + j;
    int oc = t * 16 + (lane & 15);
    int lwch = (K < 16) ? K : (K + 16);            // skip the P0-ones block (ch 16..31)
    wpack[e] = f2bf(lw[(oc * 80 + lwch) * 9 + tap]);
  } else if (e < NWPACK + NWB) {
    int q   = e - NWPACK;
    int oc  = q & 31;
    int cse = q >> 5;                              // 0..8
    int chh = cse / 3, cww = cse % 3;              // 0=full, 1=drop first, 2=drop last
    int khlo = (chh == 1) ? 1 : 0, khhi = (chh == 2) ? 1 : 2;
    int kwlo = (cww == 1) ? 1 : 0, kwhi = (cww == 2) ? 1 : 2;
    float s = 0.f;
    for (int kh = khlo; kh <= khhi; ++kh)
      for (int kw = kwlo; kw <= kwhi; ++kw)
        for (int c = 0; c < 16; ++c)
          s += lw[(oc * 80 + 16 + c) * 9 + kh * 3 + kw];
    wb[q] = s;
  }
}

// ---- kernel 2: per-group global min/max of xg ----
__global__ __launch_bounds__(256) void k_minmax(const float* __restrict__ x,
                                                unsigned* __restrict__ mm) {
  __shared__ float smn[256], smx[256];
  int g = blockIdx.y, tid = threadIdx.x;
  float mn = 3.4e38f, mx = -3.4e38f;
  const int NG = BATCH * IN_G * HW * HW;           // 4,194,304 per group
  for (int i = blockIdx.x * 256 + tid; i < NG; i += gridDim.x * 256) {
    int b   = i >> 16;
    int c   = (i >> 12) & 15;
    int off = i & 4095;
    float v = x[((b * IN_DIM + g * IN_G + c) << 12) + off];
    mn = fminf(mn, v); mx = fmaxf(mx, v);
  }
  smn[tid] = mn; smx[tid] = mx; __syncthreads();
  for (int st = 128; st > 0; st >>= 1) {
    if (tid < st) { smn[tid] = fminf(smn[tid], smn[tid + st]);
                    smx[tid] = fmaxf(smx[tid], smx[tid + st]); }
    __syncthreads();
  }
  if (tid == 0) {
    atomicMin(&mm[g * 2 + 0], fkey(smn[0]));
    atomicMax(&mm[g * 2 + 1], fkey(smx[0]));
  }
}

// ---- kernel 3: fused basis + implicit-GEMM 3x3 conv via bf16 WMMA ----
// block = (g, b, 2 output rows), 128 threads = 4 waves; each wave: 32 px x 32 oc
// (4 WMMA tiles: every A and every B fragment feeds two WMMAs). K = 64 exactly
// (2 K-steps of 32, no padding). P0-ones contribution added as epilogue bias.
__global__ __launch_bounds__(128, 4) void k_conv(const float* __restrict__ x,
                                                 const unsigned short* __restrict__ wpack,
                                                 const float* __restrict__ wb,
                                                 const unsigned* __restrict__ mm,
                                                 float* __restrict__ y) {
  __shared__ __align__(16) float          sX[4 * IN_G * HW];   // 16 KiB raw x patch
  __shared__ __align__(16) unsigned short sBas[4 * 66 * RCH];  // 33 KiB basis halo tile
  int bid = blockIdx.x;
  int ht = bid & 31, b = (bid >> 5) & 63, g = bid >> 11;
  int h0 = ht * 2;
  int tid = threadIdx.x;

  // ---- async-DMA raw x rows h0-1..h0+2 into LDS (16 B per lane per issue) ----
  unsigned long long sbase = (unsigned long long)(x + (size_t)b * IN_DIM * HW * HW);
  #pragma unroll
  for (int k = 0; k < 8; ++k) {
    int u = tid + k * 128;                 // 1024 x 16B transfer units
    int unit = u & 15;
    int c    = (u >> 4) & 15;
    int r    = u >> 8;
    int hi = h0 - 1 + r;
    if (hi >= 0 && hi < HW) {
      unsigned lds  = (unsigned)(size_t)&sX[(r * IN_G + c) * HW + unit * 4];
      unsigned voff = (unsigned)((((g * IN_G + c) * HW * HW) + hi * HW) * 4 + unit * 16);
      asm volatile("global_load_async_to_lds_b128 %0, %1, %2"
                   :: "v"(lds), "v"(voff), "s"(sbase) : "memory");
    }
  }
  asm volatile("s_wait_asynccnt 0x0" ::: "memory");
  __syncthreads();

  float xmin = funkey(mm[g * 2 + 0]);
  float xmax = funkey(mm[g * 2 + 1]);
  float sc = 2.f / (xmax - xmin);

  // ---- cooperative basis build from LDS: rows h0-1..h0+2, cols -1..64, 64 ch ----
  for (int i = tid; i < 4 * 66 * IN_G; i += 128) {
    int col = i % 66;
    int c   = (i / 66) & 15;
    int r   = i / (66 * 16);
    int hi = h0 - 1 + r, wi = col - 1;
    bool inb = (hi >= 0) && (hi < HW) && (wi >= 0) && (wi < HW);
    unsigned short s0 = 0, s1 = 0, s2 = 0, s3 = 0;
    if (inb) {
      float v  = sX[(r * IN_G + c) * HW + wi];
      float xn = (v - xmin) * sc - 1.f;
      float p2 = 1.5f * xn * xn - 0.5f;
      float p3 = (2.5f * xn * xn - 1.5f) * xn;
      s0 = f2bf(silu(v)); s1 = f2bf(xn); s2 = f2bf(p2); s3 = f2bf(p3);
    }
    unsigned base = (r * 66 + col) * RCH;
    sBas[base +      c] = s0;   // silu(x)           (weights lw ch 0..15)
    sBas[base + 16 + c] = s1;   // P1 = x_norm       (weights lw ch 32..47)
    sBas[base + 32 + c] = s2;   // P2                (weights lw ch 48..63)
    sBas[base + 48 + c] = s3;   // P3                (weights lw ch 64..79)
  }
  __syncthreads();

  int wave = tid >> 5, lane = tid & 31;
  int r  = wave >> 1;            // output row within block (0..1)
  int w0 = (wave & 1) << 5;      // 32-px column half-row
  int lhalf = (lane & 16) ? 1 : 0;
  int l15   = lane & 15;

  v8f acc00 = {}, acc01 = {}, acc10 = {}, acc11 = {};   // [px tile][oc tile]
  #pragma unroll
  for (int tap = 0; tap < 9; ++tap) {
    int kh = tap / 3, kw = tap % 3;
    const unsigned short* pA0 = &sBas[((r + kh) * 66 + (w0 + l15 + kw)) * RCH];
    const unsigned short* pA1 = pA0 + 16 * RCH;          // px tile +16 columns
    #pragma unroll
    for (int s = 0; s < 2; ++s) {
      // lanes<16 hold K{0-15}, lanes>=16 hold K{16-31}
      int ch0 = s * 32 + (lhalf ? 16 : 0);
      v16bf a0 = mkfrag(*(const v8us*)(pA0 + ch0), *(const v8us*)(pA0 + ch0 + 8));
      v16bf a1 = mkfrag(*(const v8us*)(pA1 + ch0), *(const v8us*)(pA1 + ch0 + 8));
      v16us b0 = *(const v16us*)(wpack + (((0 * 9 + tap) * 2 + s) * 32 + lane) * 16);
      v16us b1 = *(const v16us*)(wpack + (((1 * 9 + tap) * 2 + s) * 32 + lane) * 16);
      v16bf bb0 = __builtin_bit_cast(v16bf, b0);
      v16bf bb1 = __builtin_bit_cast(v16bf, b1);
      acc00 = __builtin_amdgcn_wmma_f32_16x16x32_bf16(false, a0, false, bb0, (short)0, acc00, false, false);
      acc01 = __builtin_amdgcn_wmma_f32_16x16x32_bf16(false, a0, false, bb1, (short)0, acc01, false, false);
      acc10 = __builtin_amdgcn_wmma_f32_16x16x32_bf16(false, a1, false, bb0, (short)0, acc10, false, false);
      acc11 = __builtin_amdgcn_wmma_f32_16x16x32_bf16(false, a1, false, bb1, (short)0, acc11, false, false);
    }
  }

  // ---- epilogue: add the P0-ones bias (border-case table) and store ----
  // C/D layout: VGPR rI -> M = rI + 8*lhalf (pixel), lane%16 -> N (oc).
  int h = h0 + r, wp0 = w0 + l15, wp1 = wp0 + 16;
  int chh = (h == 0) ? 1 : ((h == HW - 1) ? 2 : 0);
  int cw0 = (wp0 == 0) ? 1 : 0;            // wp0 in [0,47]: only left edge possible
  int cw1 = (wp1 == HW - 1) ? 2 : 0;       // wp1 in [16,63]: only right edge possible
  const float* wb0 = wb + (chh * 3 + cw0) * OUT_G + lhalf * 8;
  const float* wb1 = wb + (chh * 3 + cw1) * OUT_G + lhalf * 8;
  float* yb = y + (((size_t)(b * OUT_DIM + g * OUT_G + lhalf * 8) * HW + h) * HW
                   + (w0 + l15));
  #pragma unroll
  for (int rI = 0; rI < 8; ++rI) {
    yb[rI * (HW * HW)]                       = acc00[rI] + wb0[rI];
    yb[rI * (HW * HW) + 16 * (HW * HW)]      = acc01[rI] + wb0[rI + 16];
    yb[rI * (HW * HW) + 16]                  = acc10[rI] + wb1[rI];
    yb[rI * (HW * HW) + 16 * (HW * HW) + 16] = acc11[rI] + wb1[rI + 16];
  }
}

// ---- kernel 4: instance norm (population var) + silu, one block per (b, oc) plane ----
__global__ __launch_bounds__(256) void k_norm(const float* __restrict__ y,
                                              float* __restrict__ out) {
  __shared__ float sS[256], sQ[256];
  int p = blockIdx.x, tid = threadIdx.x;
  const float* yp = y + (size_t)p * 4096;
  float v[16], s = 0.f, q = 0.f;
  #pragma unroll
  for (int k = 0; k < 16; ++k) { float t = yp[tid + k * 256]; v[k] = t; s += t; q += t * t; }
  sS[tid] = s; sQ[tid] = q; __syncthreads();
  for (int st = 128; st > 0; st >>= 1) {
    if (tid < st) { sS[tid] += sS[tid + st]; sQ[tid] += sQ[tid + st]; }
    __syncthreads();
  }
  float mean = sS[0] * (1.f / 4096.f);
  float var  = sQ[0] * (1.f / 4096.f) - mean * mean;
  float inv  = rsqrtf(var + 1e-5f);
  float* op = out + (size_t)p * 4096;
  #pragma unroll
  for (int k = 0; k < 16; ++k) {
    float t = (v[k] - mean) * inv;
    op[tid + k * 256] = silu(t);
  }
}

extern "C" void kernel_launch(void* const* d_in, const int* in_sizes, int n_in,
                              void* d_out, int out_size, void* d_ws, size_t ws_size,
                              hipStream_t stream) {
  const float* x  = (const float*)d_in[0];      // (64,64,64,64) f32
  const float* lw = (const float*)d_in[1];      // (32,80,3,3)   f32
  float* out = (float*)d_out;                   // (64,128,64,64) f32
  char* ws = (char*)d_ws;

  float* yws = (float*)ws;                                        // 128 MiB f32 y
  size_t MMOFF = (size_t)BATCH * OUT_DIM * HW * HW * sizeof(float);
  unsigned* mm = (unsigned*)(ws + MMOFF);                         // 8 u32 min/max keys
  unsigned short* wpack = (unsigned short*)(ws + MMOFF + 256);    // 36,864 B bf16 weights
  float* wb = (float*)(ws + MMOFF + 256 + NWPACK * 2);            // 1,152 B bias table

  k_init<<<(NWPACK + NWB + 255) / 256, 256, 0, stream>>>(lw, mm, wpack, wb);
  dim3 gmm(1024, NGROUPS);
  k_minmax<<<gmm, 256, 0, stream>>>(x, mm);
  k_conv<<<NGROUPS * BATCH * (HW / 2), 128, 0, stream>>>(x, wpack, wb, mm, yws);
  k_norm<<<BATCH * OUT_DIM, 256, 0, stream>>>(yws, out);
}
